// SemanticRelationLite_73134703116573
// MI455X (gfx1250) — matrix operand order repaired and checked by reading, not compiled, not run on hardware
//
#include <hip/hip_runtime.h>
#include <hip/hip_bf16.h>

#define BATCH 4
#define CH    512
#define NPIX  4096
#define INTER 16
#define QSTR  32   // q/k inner dim padded 16 -> 32 so one K=32 WMMA covers it

typedef __attribute__((ext_vector_type(16))) __bf16 v16bf;
typedef __attribute__((ext_vector_type(8)))  float  v8f;

union BF16Tile { uint4 u4[2]; v16bf v; };

// f32x2 -> packed bf16x2, round-half-up then v_perm_b32 byte-select (3 VALU ops)
__device__ __forceinline__ unsigned int pk2(float lo, float hi) {
  unsigned int ul = __builtin_bit_cast(unsigned int, lo) + 0x8000u;
  unsigned int uh = __builtin_bit_cast(unsigned int, hi) + 0x8000u;
  return __builtin_amdgcn_perm(uh, ul, 0x07060302u);  // {uh[31:16], ul[31:16]}
}
__device__ __forceinline__ unsigned short f2bf1(float f) {
  return (unsigned short)((__builtin_bit_cast(unsigned int, f) + 0x8000u) >> 16);
}
__device__ __forceinline__ float bf2f(unsigned short h) {
  return __builtin_bit_cast(float, ((unsigned int)h) << 16);
}

// ---------------------------------------------------------------------------
// Kernel W: f32 -> bf16 bulk convert (for the small weight matrices)
// ---------------------------------------------------------------------------
__global__ void __launch_bounds__(256)
cvt_f32_bf16(const float* __restrict__ src, unsigned short* __restrict__ dst, int n) {
  const int i = (blockIdx.x * 256 + threadIdx.x) * 4;
  if (i < n) {
    float4 f = *(const float4*)(src + i);
    *(uint2*)(dst + i) = make_uint2(pk2(f.x, f.y), pk2(f.z, f.w));
  }
}

// ---------------------------------------------------------------------------
// Kernel 0: x [B][C][N] f32 -> xT [B][N][C] bf16 (LDS tile transpose, packed stores)
// ---------------------------------------------------------------------------
__global__ void __launch_bounds__(256)
transpose_cvt(const float* __restrict__ x, unsigned short* __restrict__ xT) {
  __shared__ float tile[32][33];
  const int b  = blockIdx.z;
  const int c0 = blockIdx.y * 32;
  const int n0 = blockIdx.x * 32;
  for (int i = threadIdx.y; i < 32; i += 8)
    tile[i][threadIdx.x] = x[((size_t)(b * CH + c0 + i)) * NPIX + n0 + threadIdx.x];
  __syncthreads();
  const int tx2   = threadIdx.x & 15;          // which bf16-pair column
  const int ihalf = (threadIdx.x >> 4) * 16;   // row half
  for (int i = threadIdx.y; i < 16; i += 8) {
    const int row = ihalf + i;
    const unsigned int v = pk2(tile[2 * tx2][row], tile[2 * tx2 + 1][row]);
    *(unsigned int*)(xT + ((size_t)b * NPIX + n0 + row) * CH + c0 + 2 * tx2) = v;
  }
}

// ---------------------------------------------------------------------------
// Kernel 1: out = (xT[b][n][:] . W[o][:] + bias[o]) * scale  via bf16 WMMA,
// reduction over C=512, one wave per 16x16 tile; output layout set by strides.
// Inner loop is pure b128 loads + wmma (weights pre-converted to bf16).
// A (xT): lane<16 rows, K [0..16); lanes>=16: K [16..32).
// B (W^T): lane = column o, 16 contiguous K (W row-major along C).
// ---------------------------------------------------------------------------
__global__ void __launch_bounds__(256)
gemm_bf16_wmma(const unsigned short* __restrict__ xT,
               const unsigned short* __restrict__ Wbf,
               const float* __restrict__ bias,
               unsigned short* __restrict__ outw,
               int O, long sO, long sN, long sB, float scale, int pad)
{
  const int lane  = threadIdx.x & 31;
  const int wv    = threadIdx.x >> 5;
  const int b     = blockIdx.z;
  const int n0    = blockIdx.x * 16;
  const int otile = blockIdx.y * 8 + wv;
  if (otile * 16 >= O) return;          // wave-uniform: EXEC stays all-ones for WMMA
  const int o0 = otile * 16;

  const int rowA = lane & 15;
  const int offA = (lane < 16) ? 0 : 16;
  const int colB = lane & 15;
  const int krB  = (lane >> 4) << 4;

  const unsigned short* ap = xT  + ((size_t)b * NPIX + n0 + rowA) * CH + offA;
  const unsigned short* wp = Wbf + (size_t)(o0 + colB) * CH + krB;

  v8f acc = {0.f, 0.f, 0.f, 0.f, 0.f, 0.f, 0.f, 0.f};
#pragma unroll 4
  for (int k0 = 0; k0 < CH; k0 += 32) {
    BF16Tile A, Bm;
    A.u4[0]  = *(const uint4*)(ap + k0);
    A.u4[1]  = *(const uint4*)(ap + k0 + 8);
    Bm.u4[0] = *(const uint4*)(wp + k0);
    Bm.u4[1] = *(const uint4*)(wp + k0 + 8);
    acc = __builtin_amdgcn_wmma_f32_16x16x32_bf16(false, A.v, false, Bm.v,
                                                  (short)0, acc, false, false);
  }

  const int   o     = o0 + colB;
  const float bval  = bias[o];
  const int   rhalf = (lane >> 4) * 8;
  unsigned short* ob = outw + (size_t)b * sB + (size_t)o * sO + (size_t)(n0 + rhalf) * sN;
#pragma unroll
  for (int r = 0; r < 8; ++r) {
    ob[(size_t)r * sN] = f2bf1((acc[r] + bval) * scale);
    if (pad)  // zero-fill padded K columns 16..31 for the q/k buffers
      ob[(size_t)16 * sO + (size_t)r * sN] = 0;
  }
}

// ---------------------------------------------------------------------------
// Kernel 2: attention for one (batch, 16-query tile).
//   Pass A: S[16][4096] = q.k^T via WMMA into LDS bf16 (128 KB of the 320KB WGP LDS)
//   Pass B: rowmax -> exp -> rowsum (normalization deferred to epilogue)
//   Pass C: out^T[16][512] = P @ V^T via WMMA (A from LDS, B streamed from L2)
// ---------------------------------------------------------------------------
__global__ void __launch_bounds__(256)
attn_wmma(const unsigned short* __restrict__ qws,   // [B][N][32] bf16, scale folded
          const unsigned short* __restrict__ kws,   // [B][N][32] bf16
          const unsigned short* __restrict__ vws,   // [B][C][N]  bf16
          const float* __restrict__ gammaP,
          float* __restrict__ out)                  // [B][C][N]  f32
{
  extern __shared__ unsigned char smem_raw[];
  unsigned short* Sbf    = (unsigned short*)smem_raw;                     // 16*4096 bf16
  float*          red    = (float*)(smem_raw + 16 * NPIX * sizeof(unsigned short));
  float*          rowmax = red;          // 16
  float*          rowsum = red + 16;     // 16
  float*          scratch= red + 32;     // 16*16

  const int tid  = threadIdx.x;
  const int lane = tid & 31;
  const int wv   = tid >> 5;
  const int b    = blockIdx.y;
  const int n0   = blockIdx.x * 16;

  const int rowA  = lane & 15;
  const int offA  = (lane < 16) ? 0 : 16;
  const int colB  = lane & 15;
  const int krB   = (lane >> 4) << 4;
  const int rhalf = (lane >> 4) * 8;

  // ---- Q tile (A operand), identical in every wave ----
  BF16Tile Aq;
  {
    const size_t qb = ((size_t)b * NPIX + n0 + rowA) * QSTR;
    Aq.u4[0] = *(const uint4*)(qws + qb + offA);
    Aq.u4[1] = *(const uint4*)(qws + qb + 8 + offA);
  }

  // ---- Pass A: logits into LDS ----
  for (int mt = wv; mt < NPIX / 16; mt += 8) {
    const int m0 = mt * 16;
    BF16Tile Bk;
    const size_t kb = ((size_t)b * NPIX + m0 + colB) * QSTR + krB;
    Bk.u4[0] = *(const uint4*)(kws + kb);
    Bk.u4[1] = *(const uint4*)(kws + kb + 8);
    v8f s = {0.f, 0.f, 0.f, 0.f, 0.f, 0.f, 0.f, 0.f};
    s = __builtin_amdgcn_wmma_f32_16x16x32_bf16(false, Aq.v, false, Bk.v,
                                                (short)0, s, false, false);
#pragma unroll
    for (int r = 0; r < 8; ++r)
      Sbf[(rhalf + r) * NPIX + m0 + colB] = f2bf1(s[r]);
  }
  __syncthreads();

  // ---- Pass B: rowmax, exp (unnormalized), rowsum ----
  {
    const int r = tid >> 4;   // row 0..15
    const int g = tid & 15;   // 16 threads cooperate per row
    float lmax = -3.0e38f;
    for (int i = 0; i < NPIX / 16; ++i)
      lmax = fmaxf(lmax, bf2f(Sbf[r * NPIX + g + (i << 4)]));
    scratch[r * 16 + g] = lmax;
    __syncthreads();
    if (tid < 16) {
      float m = scratch[tid * 16];
      for (int j = 1; j < 16; ++j) m = fmaxf(m, scratch[tid * 16 + j]);
      rowmax[tid] = m;
    }
    __syncthreads();
    const float rmax = rowmax[r];
    float lsum = 0.f;
    for (int i = 0; i < NPIX / 16; ++i) {
      const int idx = r * NPIX + g + (i << 4);
      const float e = __expf(bf2f(Sbf[idx]) - rmax);
      Sbf[idx] = f2bf1(e);
      lsum += e;
    }
    scratch[r * 16 + g] = lsum;
    __syncthreads();
    if (tid < 16) {
      float su = 0.f;
      for (int j = 0; j < 16; ++j) su += scratch[tid * 16 + j];
      rowsum[tid] = su;
    }
    __syncthreads();
  }

  // ---- Pass C: out^T = P @ V^T ; wave wv owns channels [wv*64, wv*64+64) ----
  const int c0 = wv * 64;
  const unsigned short* vbase =
      vws + ((size_t)b * CH + c0 + colB) * NPIX + krB;  // + t*16*NPIX + m0
  v8f acc0 = {0.f,0.f,0.f,0.f,0.f,0.f,0.f,0.f};
  v8f acc1 = acc0, acc2 = acc0, acc3 = acc0;
#pragma unroll 2
  for (int m0 = 0; m0 < NPIX; m0 += 32) {
    BF16Tile Ap;   // P tile from LDS (16-byte aligned ds_load_b128)
    Ap.u4[0] = *(const uint4*)(Sbf + rowA * NPIX + m0 + offA);
    Ap.u4[1] = *(const uint4*)(Sbf + rowA * NPIX + m0 + 8 + offA);
#pragma unroll
    for (int t = 0; t < 4; ++t) {
      const unsigned short* vp = vbase + (size_t)t * 16 * NPIX + m0;
      __builtin_prefetch(vp + 256, 0, 1);   // global_prefetch_b8, ~8 iters ahead
      BF16Tile Bv;
      Bv.u4[0] = *(const uint4*)(vp);
      Bv.u4[1] = *(const uint4*)(vp + 8);
      v8f* accp = (t == 0) ? &acc0 : (t == 1) ? &acc1 : (t == 2) ? &acc2 : &acc3;
      *accp = __builtin_amdgcn_wmma_f32_16x16x32_bf16(false, Ap.v, false, Bv.v,
                                                      (short)0, *accp, false, false);
    }
  }

  // ---- Epilogue: scale by gamma / rowsum, write f32 [B][C][N] ----
  const float gm = gammaP[0];
#pragma unroll
  for (int t = 0; t < 4; ++t) {
    const v8f a = (t == 0) ? acc0 : (t == 1) ? acc1 : (t == 2) ? acc2 : acc3;
    const int c = c0 + t * 16 + colB;
#pragma unroll
    for (int r = 0; r < 8; ++r) {
      const int row = rhalf + r;
      out[((size_t)b * CH + c) * NPIX + n0 + row] = a[r] * gm / rowsum[row];
    }
  }
}

// ---------------------------------------------------------------------------
extern "C" void kernel_launch(void* const* d_in, const int* in_sizes, int n_in,
                              void* d_out, int out_size, void* d_ws, size_t ws_size,
                              hipStream_t stream) {
  const float* x     = (const float*)d_in[0];
  const float* Wq    = (const float*)d_in[1];
  const float* bq    = (const float*)d_in[2];
  const float* Wk    = (const float*)d_in[3];
  const float* bk    = (const float*)d_in[4];
  const float* Wv    = (const float*)d_in[5];
  const float* bv    = (const float*)d_in[6];
  const float* gamma = (const float*)d_in[7];
  float* out = (float*)d_out;

  // workspace layout (bf16): xT 16MB | q 1MB | k 1MB | v 16MB | Wq/Wk/Wv ~0.6MB
  unsigned short* xT  = (unsigned short*)d_ws;
  unsigned short* qw  = xT  + (size_t)BATCH * NPIX * CH;
  unsigned short* kw  = qw  + (size_t)BATCH * NPIX * QSTR;
  unsigned short* vw  = kw  + (size_t)BATCH * NPIX * QSTR;
  unsigned short* wqb = vw  + (size_t)BATCH * NPIX * CH;
  unsigned short* wkb = wqb + (size_t)INTER * CH;
  unsigned short* wvb = wkb + (size_t)INTER * CH;

  // weights f32 -> bf16 (tiny, one-shot)
  cvt_f32_bf16<<<dim3((INTER * CH / 4 + 255) / 256), 256, 0, stream>>>(Wq, wqb, INTER * CH);
  cvt_f32_bf16<<<dim3((INTER * CH / 4 + 255) / 256), 256, 0, stream>>>(Wk, wkb, INTER * CH);
  cvt_f32_bf16<<<dim3((CH * CH / 4 + 255) / 256), 256, 0, stream>>>(Wv, wvb, CH * CH);

  transpose_cvt<<<dim3(NPIX / 32, CH / 32, BATCH), dim3(32, 8), 0, stream>>>(x, xT);

  // q: scale 1/sqrt(16) folded; q/k stored [b][n][32] (pad cols zeroed)
  gemm_bf16_wmma<<<dim3(NPIX / 16, 1, BATCH), 256, 0, stream>>>(
      xT, wqb, bq, qw, INTER, 1L, (long)QSTR, (long)NPIX * QSTR, 0.25f, 1);
  gemm_bf16_wmma<<<dim3(NPIX / 16, 1, BATCH), 256, 0, stream>>>(
      xT, wkb, bk, kw, INTER, 1L, (long)QSTR, (long)NPIX * QSTR, 1.0f, 1);
  // v stored channel-major [b][c][n] so PV^T B-operand is K-contiguous
  gemm_bf16_wmma<<<dim3(NPIX / 16, CH / 128, BATCH), 256, 0, stream>>>(
      xT, wvb, bv, vw, CH, (long)NPIX, 1L, (long)CH * NPIX, 1.0f, 0);

  const size_t smem = (size_t)16 * NPIX * sizeof(unsigned short) + (32 + 256) * sizeof(float);
  hipFuncSetAttribute((const void*)attn_wmma,
                      hipFuncAttributeMaxDynamicSharedMemorySize, (int)smem);
  attn_wmma<<<dim3(NPIX / 16, BATCH), 256, smem, stream>>>(qw, kw, vw, gamma, out);
}
